// Block_55250459296225
// MI455X (gfx1250) — matrix operand order repaired and checked by reading, not compiled, not run on hardware
//
#include <hip/hip_runtime.h>
#include <math.h>

typedef __attribute__((ext_vector_type(16))) _Float16 v16h;
typedef __attribute__((ext_vector_type(8)))  _Float16 v8h;
typedef __attribute__((ext_vector_type(8)))  float    v8f;

#define EPSBN 1e-5f
constexpr int Npts = 32768;
constexpr int DIMC = 192;
constexpr int HIDC = 384;
constexpr int HALFC = 96;
constexpr int KNN  = 16;
constexpr int MAXCIN = 384;
constexpr int LFA_WAVES = 8;
constexpr int LFA_BLOCKS = 1024;

// ---------------------------------------------------------------- helpers ---

__device__ __forceinline__ v8f wmma16(v16h a, v16h b, v8f c) {
  // v_wmma_f32_16x16x32_f16: D = A(16x32 f16) * B(32x16 f16) + C(16x16 f32)
  return __builtin_amdgcn_wmma_f32_16x16x32_f16(false, a, false, b, (short)0, c,
                                                false, false);
}

// A fragment (16x32 f16). Per ISA layout: lane<16 holds K[kt..kt+7],K[kt+16..kt+23];
// lane>=16 holds K[kt+8..kt+15],K[kt+24..kt+31]. Caller passes &row[kt + (lane<16?0:8)].
__device__ __forceinline__ v16h load_a_frag(const _Float16* p) {
  v8h lo = *(const v8h*)p;
  v8h hi = *(const v8h*)(p + 16);
  v16h a;
#pragma unroll
  for (int h = 0; h < 8; ++h) { a[h] = lo[h]; a[h + 8] = hi[h]; }
  return a;
}

// B fragment (32x16 f16) from a column-major slab Wt[col][k].
// lane gives col=lane&15; lane<16 covers K[kt..kt+7],K[kt+16..kt+23]; lane>=16 K[kt+8..kt+15],K[kt+24..kt+31].
__device__ __forceinline__ v16h load_b_frag(const _Float16* p) {
  v8h lo = *(const v8h*)p;
  v8h hi = *(const v8h*)(p + 16);
  v16h b;
#pragma unroll
  for (int h = 0; h < 8; ++h) { b[h] = lo[h]; b[h + 8] = hi[h]; }
  return b;
}

__device__ __forceinline__ float gelu_exact(float v) {
  return 0.5f * v * (1.0f + erff(v * 0.70710678118654752f));
}

// ------------------------------------------------------- fused dense GEMM ---
// out[N,Cout] = post( A[N,Cin] @ W[Cin,Cout] )
// post: (+bias) -> (gelu if act) -> (BN if bnw) -> (+res if res)
// 4 waves per block, 64 rows per block, weights staged transposed in LDS.

__global__ __launch_bounds__(128) void gemm_fused(
    const float* __restrict__ A, const float* __restrict__ W,
    const float* __restrict__ bias, const float* __restrict__ res,
    const float* __restrict__ bnw, const float* __restrict__ bnb,
    const float* __restrict__ bnrm, const float* __restrict__ bnrv,
    float* __restrict__ out, int Cin, int Cout, int act) {
  __shared__ __align__(16) _Float16 As[64][MAXCIN];   // 48 KB
  __shared__ __align__(16) _Float16 Wt[16][MAXCIN];   // 12 KB (transposed slab)

  const int tid  = threadIdx.x;
  const int lane = tid & 31;
  const int wave = tid >> 5;
  const int rowBase = blockIdx.x * 64;

  // stage activations (f32 -> f16) into LDS
  for (int e = tid; e < 64 * Cin; e += 128) {
    int r = e / Cin, c = e % Cin;
    As[r][c] = (_Float16)A[(size_t)(rowBase + r) * Cin + c];
  }

  const int nCT = Cout >> 4;
  for (int ct = 0; ct < nCT; ++ct) {
    __syncthreads();
    // stage W[:, ct*16 .. +16) transposed: Wt[n][k]
    for (int e = tid; e < 16 * Cin; e += 128) {
      int n = e / Cin, k = e % Cin;
      Wt[n][k] = (_Float16)W[(size_t)k * Cout + ct * 16 + n];
    }
    if (ct + 1 < nCT) __builtin_prefetch(&W[(size_t)(ct + 1) * 16], 0, 0);
    __syncthreads();

    v8f acc;
#pragma unroll
    for (int r = 0; r < 8; ++r) acc[r] = 0.0f;

    const int m = lane & 15;
    const int aOff = (lane < 16) ? 0 : 8;
    const int bOff = (lane < 16) ? 0 : 8;
    for (int kt = 0; kt < Cin; kt += 32) {
      v16h a = load_a_frag(&As[wave * 16 + m][kt + aOff]);
      v16h b = load_b_frag(&Wt[m][kt + bOff]);
      acc = wmma16(a, b, acc);
    }

    // epilogue
    const int col = ct * 16 + m;
    float bv = bias ? bias[col] : 0.0f;
    float scale = 1.0f, shift = 0.0f;
    if (bnw) {
      float inv = rsqrtf(bnrv[col] + EPSBN) * bnw[col];
      scale = inv;
      shift = bnb[col] - bnrm[col] * inv;
    }
#pragma unroll
    for (int r = 0; r < 8; ++r) {
      int row = rowBase + wave * 16 + r + ((lane < 16) ? 0 : 8);
      float v = acc[r] + bv;
      if (act) v = gelu_exact(v);
      if (bnw) v = v * scale + shift;
      if (res) v += res[(size_t)row * Cout + col];
      out[(size_t)row * Cout + col] = v;
    }
  }
}

// ------------------------------------------------------ fused LFA kernel ---
// Persistent-weight version: all three NCA weight matrices live transposed in
// LDS for the whole kernel (166 KB of the WGP's 320 KB); one barrier total.
// One wave per point (16 neighbors == one 16-row WMMA tile); waves grid-stride
// over points with no further synchronization (tile buffers are wave-private).
// p0 = xyzs@nca1+b ; p_local = max_k p0 ; p1 = p0@nca2+b ; p2=[p1|p_local]
// t  = gelu(p2@nca3a+b) ; pe = t@nca3b+b ; out = x + BN(max_k(xs + pe))

__global__ __launch_bounds__(256) void lfa_fused(
    const float* __restrict__ xyz, const int* __restrict__ knn,
    const float* __restrict__ xp, const float* __restrict__ x_in,
    const float* __restrict__ w1, const float* __restrict__ b1,
    const float* __restrict__ w2, const float* __restrict__ b2,
    const float* __restrict__ w3a, const float* __restrict__ b3a,
    const float* __restrict__ w3b, const float* __restrict__ b3b,
    const float* __restrict__ bnw, const float* __restrict__ bnb,
    const float* __restrict__ bnrm, const float* __restrict__ bnrv,
    float* __restrict__ x_out) {
  __shared__ __align__(16) _Float16 W2t[HALFC][HALFC];          // 18 KB
  __shared__ __align__(16) _Float16 W3at[DIMC][DIMC];           // 72 KB
  __shared__ __align__(16) _Float16 W3bt[DIMC][DIMC];           // 72 KB
  __shared__ __align__(16) _Float16 As[LFA_WAVES][16][DIMC];    // 48 KB (p2)
  __shared__ __align__(16) _Float16 Ts[LFA_WAVES][16][DIMC];    // 48 KB (p0/t)
  __shared__ int   idxS[LFA_WAVES][KNN];
  __shared__ float rel[LFA_WAVES][KNN][3];

  const int tid  = threadIdx.x;
  const int lane = tid & 31;
  const int wave = tid >> 5;

  // ---- stage all weights transposed (col-major slabs) ONCE --------------
  for (int e = tid; e < HALFC * HALFC; e += 256) {
    int n = e / HALFC, k = e % HALFC;
    W2t[n][k] = (_Float16)w2[(size_t)k * HALFC + n];
  }
  for (int e = tid; e < DIMC * DIMC; e += 256) {
    int n = e / DIMC, k = e % DIMC;
    W3at[n][k] = (_Float16)w3a[(size_t)k * DIMC + n];
    W3bt[n][k] = (_Float16)w3b[(size_t)k * DIMC + n];
  }
  __syncthreads();   // the only block-wide barrier

  const int m    = lane & 15;
  const int aOff = (lane < 16) ? 0 : 8;
  const int bOff = (lane < 16) ? 0 : 8;
  const int rowH = (lane < 16) ? 0 : 8;

  for (int p = blockIdx.x * LFA_WAVES + wave; p < Npts;
       p += gridDim.x * LFA_WAVES) {
    // ---- neighbor indices + relative coords (wave-private LDS) ----------
    if (lane < KNN) {
      int id = knn[(size_t)p * KNN + lane];
      idxS[wave][lane] = id;
#pragma unroll
      for (int d = 0; d < 3; ++d)
        rel[wave][lane][d] = xyz[(size_t)id * 3 + d] - xyz[(size_t)p * 3 + d];
    }

    // ---- p0 = xyzs @ nca1_w + b (K=3 scalar FMA) -> Ts[wave][k][c<96] ---
    for (int e = lane; e < KNN * HALFC; e += 32) {
      int k = e / HALFC, c = e % HALFC;
      float s = b1[c];
#pragma unroll
      for (int d = 0; d < 3; ++d) s += rel[wave][k][d] * w1[d * HALFC + c];
      Ts[wave][k][c] = (_Float16)s;
    }

    // ---- p_local = max_k p0 -> broadcast into As cols 96..191 -----------
    for (int c = lane; c < HALFC; c += 32) {
      float mx = -3.4e38f;
#pragma unroll
      for (int k = 0; k < KNN; ++k) mx = fmaxf(mx, (float)Ts[wave][k][c]);
      _Float16 h = (_Float16)mx;
#pragma unroll
      for (int k = 0; k < KNN; ++k) As[wave][k][HALFC + c] = h;
    }

    // ---- p1 = p0 @ nca2_w + b2 -> As cols 0..95 -------------------------
    for (int ct = 0; ct < HALFC / 16; ++ct) {
      v8f acc;
#pragma unroll
      for (int r = 0; r < 8; ++r) acc[r] = 0.0f;
#pragma unroll
      for (int kt = 0; kt < HALFC; kt += 32) {
        v16h a = load_a_frag(&Ts[wave][m][kt + aOff]);
        v16h b = load_b_frag(&W2t[ct * 16 + m][kt + bOff]);
        acc = wmma16(a, b, acc);
      }
      int col = ct * 16 + m;
      float bv = b2[col];
#pragma unroll
      for (int r = 0; r < 8; ++r)
        As[wave][r + rowH][col] = (_Float16)(acc[r] + bv);
    }

    // ---- t = gelu(p2 @ nca3a + b3a) -> Ts (overwrites p0) ---------------
    for (int ct = 0; ct < DIMC / 16; ++ct) {
      v8f acc;
#pragma unroll
      for (int r = 0; r < 8; ++r) acc[r] = 0.0f;
#pragma unroll
      for (int kt = 0; kt < DIMC; kt += 32) {
        v16h a = load_a_frag(&As[wave][m][kt + aOff]);
        v16h b = load_b_frag(&W3at[ct * 16 + m][kt + bOff]);
        acc = wmma16(a, b, acc);
      }
      int col = ct * 16 + m;
      float bv = b3a[col];
#pragma unroll
      for (int r = 0; r < 8; ++r)
        Ts[wave][r + rowH][col] = (_Float16)gelu_exact(acc[r] + bv);
    }

    // ---- pe = t @ nca3b + b3b ; fuse gather(xs) + max_k + BN + residual -
    const float* xpp = xp + (size_t)p * DIMC;
    for (int ct = 0; ct < DIMC / 16; ++ct) {
      v8f acc;
#pragma unroll
      for (int r = 0; r < 8; ++r) acc[r] = 0.0f;
#pragma unroll
      for (int kt = 0; kt < DIMC; kt += 32) {
        v16h a = load_a_frag(&Ts[wave][m][kt + aOff]);
        v16h b = load_b_frag(&W3bt[ct * 16 + m][kt + bOff]);
        acc = wmma16(a, b, acc);
      }
      int col = ct * 16 + m;
      float bv  = b3b[col];
      float xpc = xpp[col];
      float mx  = -3.4e38f;
#pragma unroll
      for (int r = 0; r < 8; ++r) {
        int k  = r + rowH;                            // neighbor row of tile
        int id = idxS[wave][k];
        float xs = xp[(size_t)id * DIMC + col] - xpc; // gather + subtract
        mx = fmaxf(mx, acc[r] + bv + xs);
      }
      mx = fmaxf(mx, __shfl_xor(mx, 16, 32));         // other row-half
      if (lane < 16) {
        float inv = rsqrtf(bnrv[col] + EPSBN) * bnw[col];
        float v = (mx - bnrm[col]) * inv + bnb[col];
        x_out[(size_t)p * DIMC + col] = x_in[(size_t)p * DIMC + col] + v;
      }
    }
  }
}

// ------------------------------------------------------------------ host ---

extern "C" void kernel_launch(void* const* d_in, const int* in_sizes, int n_in,
                              void* d_out, int out_size, void* d_ws, size_t ws_size,
                              hipStream_t stream) {
  const float* xyz    = (const float*)d_in[0];
  const float* x0     = (const float*)d_in[1];
  const int*   knn    = (const int*)d_in[2];
  const float* g_pos  = (const float*)d_in[3];
  const float* gpe_w  = (const float*)d_in[4];
  const float* m0w1   = (const float*)d_in[5];
  const float* m0b1   = (const float*)d_in[6];
  const float* m0w2   = (const float*)d_in[7];
  const float* m0bnw  = (const float*)d_in[8];
  const float* m0bnb  = (const float*)d_in[9];
  const float* m0bnrm = (const float*)d_in[10];
  const float* m0bnrv = (const float*)d_in[11];
  const float* proj   = (const float*)d_in[12];
  const float* n1w    = (const float*)d_in[13];
  const float* n1b    = (const float*)d_in[14];
  const float* n2w    = (const float*)d_in[15];
  const float* n2b    = (const float*)d_in[16];
  const float* n3aw   = (const float*)d_in[17];
  const float* n3ab   = (const float*)d_in[18];
  const float* n3bw   = (const float*)d_in[19];
  const float* n3bb   = (const float*)d_in[20];
  const float* lbnw   = (const float*)d_in[21];
  const float* lbnb   = (const float*)d_in[22];
  const float* lbnrm  = (const float*)d_in[23];
  const float* lbnrv  = (const float*)d_in[24];
  const float* mw1    = (const float*)d_in[25];
  const float* mb1    = (const float*)d_in[26];
  const float* mw2    = (const float*)d_in[27];
  const float* mbnw   = (const float*)d_in[28];
  const float* mbnb   = (const float*)d_in[29];
  const float* mbnrm  = (const float*)d_in[30];
  const float* mbnrv  = (const float*)d_in[31];
  float* out = (float*)d_out;

  // workspace layout: xa | xb | scratch(max(N*HID, N*DIM))
  float* xa = (float*)d_ws;
  float* xb = xa + (size_t)Npts * DIMC;
  float* sc = xb + (size_t)Npts * DIMC;

  const int gemmGrid = Npts / 64;  // 512 blocks x 128 threads

  // x = x0 + g_pos @ gpe_w
  gemm_fused<<<gemmGrid, 128, 0, stream>>>(g_pos, gpe_w, nullptr, x0,
      nullptr, nullptr, nullptr, nullptr, xa, 64, DIMC, 0);

  // x = x + BN(gelu(x@w1+b1)@w2)
  gemm_fused<<<gemmGrid, 128, 0, stream>>>(xa, m0w1, m0b1, nullptr,
      nullptr, nullptr, nullptr, nullptr, sc, DIMC, HIDC, 1);
  gemm_fused<<<gemmGrid, 128, 0, stream>>>(sc, m0w2, nullptr, xa,
      m0bnw, m0bnb, m0bnrm, m0bnrv, xb, HIDC, DIMC, 0);

  float* cur = xb;
  float* nxt = xa;
  for (int i = 0; i < 4; ++i) {
    // xp = x @ lfa_proj[i]
    gemm_fused<<<gemmGrid, 128, 0, stream>>>(cur, proj + (size_t)i * DIMC * DIMC,
        nullptr, nullptr, nullptr, nullptr, nullptr, nullptr, sc, DIMC, DIMC, 0);
    // x = x + BN(max_k(xs + pe))
    lfa_fused<<<LFA_BLOCKS, 256, 0, stream>>>(xyz, knn, sc, cur,
        n1w + (size_t)i * 3 * HALFC, n1b + (size_t)i * HALFC,
        n2w + (size_t)i * HALFC * HALFC, n2b + (size_t)i * HALFC,
        n3aw + (size_t)i * DIMC * DIMC, n3ab + (size_t)i * DIMC,
        n3bw + (size_t)i * DIMC * DIMC, n3bb + (size_t)i * DIMC,
        lbnw + (size_t)i * DIMC, lbnb + (size_t)i * DIMC,
        lbnrm + (size_t)i * DIMC, lbnrv + (size_t)i * DIMC, nxt);
    { float* t = cur; cur = nxt; nxt = t; }

    if (i & 1) {
      int j = i >> 1;
      gemm_fused<<<gemmGrid, 128, 0, stream>>>(cur, mw1 + (size_t)j * DIMC * HIDC,
          mb1 + (size_t)j * HIDC, nullptr, nullptr, nullptr, nullptr, nullptr,
          sc, DIMC, HIDC, 1);
      float* dst = (i == 3) ? out : nxt;
      gemm_fused<<<gemmGrid, 128, 0, stream>>>(sc, mw2 + (size_t)j * HIDC * DIMC,
          nullptr, cur, mbnw + (size_t)j * DIMC, mbnb + (size_t)j * DIMC,
          mbnrm + (size_t)j * DIMC, mbnrv + (size_t)j * DIMC, dst, HIDC, DIMC, 0);
      if (i != 3) { float* t = cur; cur = nxt; nxt = t; } else { cur = dst; }
    }
  }
}
